// MultiHeadAttention_65025804861981
// MI455X (gfx1250) — compile-verified, matrix operand order checked by reading, not run on hardware
//
#include <hip/hip_runtime.h>

// ---------------------------------------------------------------------------
// CDNA5 (gfx1250, wave32) WMMA types + feature detection
// ---------------------------------------------------------------------------
typedef __attribute__((ext_vector_type(16))) __bf16 v16bf;
typedef __attribute__((ext_vector_type(8)))  float  v8f;
typedef __attribute__((ext_vector_type(4)))  int    v4i_t;

#define AS1 __attribute__((address_space(1)))
#define AS3 __attribute__((address_space(3)))

#if defined(__has_builtin)
#  if __has_builtin(__builtin_amdgcn_global_load_async_to_lds_b128) && \
      __has_builtin(__builtin_amdgcn_s_wait_asynccnt)
#    define HAVE_ASYNC_LDS 1
#  endif
#endif
#ifndef HAVE_ASYNC_LDS
#  define HAVE_ASYNC_LDS 0
#endif

static __device__ __forceinline__ unsigned short f32_to_bf16(float f) {
    union { float f; unsigned u; } x; x.f = f;
    unsigned u = x.u + 0x7FFFu + ((x.u >> 16) & 1u);   // round-to-nearest-even
    return (unsigned short)(u >> 16);
}
static __device__ __forceinline__ float bf16_to_f32(unsigned short h) {
    union { unsigned u; float f; } x; x.u = ((unsigned)h) << 16;
    return x.f;
}

static __device__ __forceinline__ v8f wmma_bf16(v16bf a, v16bf b, v8f c) {
    // (neg_a, A, neg_b, B, c_mod, C, reuse_a, reuse_b)
    return __builtin_amdgcn_wmma_f32_16x16x32_bf16(false, a, false, b,
                                                   (short)0, c, false, false);
}

// Load a 16x32 bf16 A/B fragment from LDS tile (row stride 32 elements).
// ISA 7.12.2: lane L<16 -> row L, K {0..7,16..23}; lane>=16 -> K {8..15,24..31}
static __device__ __forceinline__ v16bf load_frag_lds(const unsigned short* tile, int lane) {
    const int row = lane & 15, half = lane >> 4;
    const unsigned short* p = tile + row * 32 + half * 8;
    union { v16bf v; uint4 q[2]; } u;
    u.q[0] = *(const uint4*)(p);
    u.q[1] = *(const uint4*)(p + 16);
    return u.v;
}

// Same fragment pattern directly from global bf16 memory, row stride ldRow elems.
static __device__ __forceinline__ v16bf load_frag_global(const unsigned short* base,
                                                         long ldRow, int lane) {
    const int row = lane & 15, half = lane >> 4;
    const unsigned short* p = base + (long)row * ldRow + half * 8;
    union { v16bf v; uint4 q[2]; } u;
    u.q[0] = *(const uint4*)(p);
    u.q[1] = *(const uint4*)(p + 16);
    return u.v;
}

// ---------------------------------------------------------------------------
// Prep kernels: f32 -> bf16 convert, and f32 [K,N] -> bf16 [N,K] transpose
// ---------------------------------------------------------------------------
__global__ void convert_f32_bf16(const float* __restrict__ in,
                                 unsigned short* __restrict__ out, long n) {
    const long i = ((long)blockIdx.x * blockDim.x + threadIdx.x) * 4;
    if (i >= n) return;
    float4 f = *(const float4*)(in + i);
    unsigned short t[4] = { f32_to_bf16(f.x), f32_to_bf16(f.y),
                            f32_to_bf16(f.z), f32_to_bf16(f.w) };
    *(uint2*)(out + i) = *(const uint2*)(t);
}

__global__ __launch_bounds__(256)
void transpose_f32_bf16(const float* __restrict__ W, unsigned short* __restrict__ WT,
                        int K, int N) {
    __shared__ float tile[32][33];
    const int n0 = blockIdx.x * 32, k0 = blockIdx.y * 32;
    const int tx = threadIdx.x, ty = threadIdx.y;
#pragma unroll
    for (int i = ty; i < 32; i += 8)
        tile[i][tx] = W[(size_t)(k0 + i) * N + n0 + tx];
    __syncthreads();
#pragma unroll
    for (int i = ty; i < 32; i += 8)
        WT[(size_t)(n0 + i) * K + k0 + tx] = f32_to_bf16(tile[tx][i]);
}

// ---------------------------------------------------------------------------
// GEMM: C[M,N] = A[M,K] * Bt[N,K]^T, both operands bf16, pre-laid-out so LDS
// staging is a pure row-major copy -> async GLOBAL_LOAD_ASYNC_TO_LDS_B128
// with double buffering and s_wait_asynccnt (fallback: sync load + ds_store).
// Block tile 128x128, 8 waves (2x4), wave tile 64x32, K-step 32.
// ---------------------------------------------------------------------------
template <bool OUT_F32>
__global__ __launch_bounds__(256)
void gemm_bf16(const unsigned short* __restrict__ A,   // [M,K] bf16
               const unsigned short* __restrict__ Bt,  // [N,K] bf16
               void* __restrict__ Cout, int M, int N, int K) {
    __shared__ unsigned short As[2][128 * 32];
    __shared__ unsigned short Bs[2][128 * 32];
    const int tid  = threadIdx.x;
    const int lane = tid & 31, wave = tid >> 5;
    const int bm = blockIdx.y * 128, bn = blockIdx.x * 128;
    const int wm = (wave >> 2) * 64;
    const int wn = (wave & 3) * 32;

    // staging mapping: thread -> (row 0..127, col-half 0/16) of a 128x32 tile
    const int srow = tid >> 1, scb = (tid & 1) * 16;
    const unsigned short* aRow = A  + (size_t)(bm + srow) * K + scb;
    const unsigned short* bRow = Bt + (size_t)(bn + srow) * K + scb;
    const int ldsOff = srow * 32 + scb;

    auto issue_tile = [&](int k0, int buf) {
#if HAVE_ASYNC_LDS
        __builtin_amdgcn_global_load_async_to_lds_b128(
            (AS1 v4i_t*)(aRow + k0),     (AS3 v4i_t*)&As[buf][ldsOff], 0, 0);
        __builtin_amdgcn_global_load_async_to_lds_b128(
            (AS1 v4i_t*)(aRow + k0 + 8), (AS3 v4i_t*)&As[buf][ldsOff + 8], 0, 0);
        __builtin_amdgcn_global_load_async_to_lds_b128(
            (AS1 v4i_t*)(bRow + k0),     (AS3 v4i_t*)&Bs[buf][ldsOff], 0, 0);
        __builtin_amdgcn_global_load_async_to_lds_b128(
            (AS1 v4i_t*)(bRow + k0 + 8), (AS3 v4i_t*)&Bs[buf][ldsOff + 8], 0, 0);
#else
        uint4 a0 = *(const uint4*)(aRow + k0);
        uint4 a1 = *(const uint4*)(aRow + k0 + 8);
        uint4 b0 = *(const uint4*)(bRow + k0);
        uint4 b1 = *(const uint4*)(bRow + k0 + 8);
        *(uint4*)(&As[buf][ldsOff])     = a0;
        *(uint4*)(&As[buf][ldsOff + 8]) = a1;
        *(uint4*)(&Bs[buf][ldsOff])     = b0;
        *(uint4*)(&Bs[buf][ldsOff + 8]) = b1;
#endif
    };

    v8f acc[4][2] = {};
    const int nk = K >> 5;

    issue_tile(0, 0);
    for (int i = 0; i < nk; ++i) {
        const int buf = i & 1;
        if (i + 1 < nk) {
            __builtin_prefetch(aRow + (i + 2) * 32, 0, 0);   // global_prefetch_b8
            __builtin_prefetch(bRow + (i + 2) * 32, 0, 0);
            issue_tile((i + 1) * 32, buf ^ 1);
#if HAVE_ASYNC_LDS
            __builtin_amdgcn_s_wait_asynccnt(4);             // tile i complete
#endif
        } else {
#if HAVE_ASYNC_LDS
            __builtin_amdgcn_s_wait_asynccnt(0);
#endif
        }
        __syncthreads();   // all waves' buffer `buf` visible

        v16bf afrag[4], bfrag[2];
#pragma unroll
        for (int a = 0; a < 4; ++a) afrag[a] = load_frag_lds(&As[buf][(wm + 16 * a) * 32], lane);
#pragma unroll
        for (int b = 0; b < 2; ++b) bfrag[b] = load_frag_lds(&Bs[buf][(wn + 16 * b) * 32], lane);
#pragma unroll
        for (int a = 0; a < 4; ++a)
#pragma unroll
            for (int b = 0; b < 2; ++b)
                acc[a][b] = wmma_bf16(afrag[a], bfrag[b], acc[a][b]);
        __syncthreads();   // done reading `buf` before it is refilled
    }

    // epilogue: C layout lane<16 -> M=r, lane>=16 -> M=8+r, N=lane&15
    const int half = lane >> 4, lcol = lane & 15;
#pragma unroll
    for (int a = 0; a < 4; ++a)
#pragma unroll
        for (int b = 0; b < 2; ++b) {
            const int r0 = bm + wm + 16 * a + half * 8;
            const int c  = bn + wn + 16 * b + lcol;
#pragma unroll
            for (int r = 0; r < 8; ++r) {
                const float v = acc[a][b][r];
                if (OUT_F32)
                    ((float*)Cout)[(size_t)(r0 + r) * N + c] = v;
                else
                    ((unsigned short*)Cout)[(size_t)(r0 + r) * N + c] = f32_to_bf16(v);
            }
        }
}

// ---------------------------------------------------------------------------
// RoPE (rotate-half, base 10000) in place on q,k slices of bf16 qkv ws.
// qkv layout: [B, S, 3*H*HD]
// ---------------------------------------------------------------------------
__global__ void rope_kernel(unsigned short* qkv, int B, int S, int H, int HD) {
    const int half = HD / 2;
    const int stride = 3 * H * HD;
    long idx = (long)blockIdx.x * blockDim.x + threadIdx.x;
    const long total = (long)B * S * H * half * 2;
    if (idx >= total) return;
    const int d  = (int)(idx % half); idx /= half;
    const int qk = (int)(idx % 2);    idx /= 2;
    const int h  = (int)(idx % H);    idx /= H;
    const int s  = (int)(idx % S);
    const long b = idx / S;

    const float inv = __powf(10000.0f, -(float)d / (float)half);
    float sn, cs;
    __sincosf((float)s * inv, &sn, &cs);

    unsigned short* base = qkv + ((b * S + s) * (size_t)stride)
                         + (size_t)qk * (H * HD) + (size_t)h * HD;
    const float x0 = bf16_to_f32(base[d]);
    const float x1 = bf16_to_f32(base[d + half]);
    base[d]        = f32_to_bf16(x0 * cs - x1 * sn);
    base[d + half] = f32_to_bf16(x1 * cs + x0 * sn);
}

// ---------------------------------------------------------------------------
// Causal flash attention. qkv: bf16 [B,S,3*H*HD]; out: bf16 [B,S,H*HD].
// Block: 256 thr (8 waves) = 128 query rows of one (b,h); wave = 16 rows.
// ---------------------------------------------------------------------------
__global__ __launch_bounds__(256)
void attn_kernel(const unsigned short* __restrict__ qkv,
                 unsigned short* __restrict__ out, int S, int H, int HD) {
    __shared__ unsigned short Vt[128 * 32];      // V^T tile: [HD=128][32 keys]
    __shared__ unsigned short Ps[8 * 16 * 32];   // per-wave P scratch [16][32]

    const int tid = threadIdx.x, lane = tid & 31, wave = tid >> 5;
    const int halfl = lane >> 4, lcol = lane & 15;
    const int h = blockIdx.y, b = blockIdx.z;
    const int stride = 3 * H * HD;               // 6144
    const int qRow0 = blockIdx.x * 128 + wave * 16;

    const size_t rowBase = ((size_t)b * S) * stride;
    const unsigned short* Qb = qkv + rowBase + (size_t)h * HD;
    const unsigned short* Kb = qkv + rowBase + (size_t)(H * HD) + (size_t)h * HD;
    const unsigned short* Vb = qkv + rowBase + (size_t)(2 * H * HD) + (size_t)h * HD;
    const float scale = rsqrtf((float)HD);

    v16bf qfrag[4];
#pragma unroll
    for (int c = 0; c < 4; ++c)
        qfrag[c] = load_frag_global(Qb + (size_t)qRow0 * stride + c * 32, stride, lane);

    v8f acc[8] = {};                 // O tile: 16 rows x 128 dims
    float mrow[8], lrow[8];
#pragma unroll
    for (int r = 0; r < 8; ++r) { mrow[r] = -1e30f; lrow[r] = 0.0f; }

    unsigned short* myP = Ps + wave * (16 * 32);
    const int nsteps = blockIdx.x * 4 + 4;       // 32-key steps up to block max

    for (int ks = 0; ks < nsteps; ++ks) {
        const int kBase = ks * 32;

        __syncthreads();   // previous iteration done reading Vt
        {   // cooperative transpose-load of V tile: Vt[d][kk] = V[kBase+kk][d]
            const int kk = tid >> 3, db = (tid & 7) * 16;
            const unsigned short* vp = Vb + (size_t)(kBase + kk) * stride + db;
            unsigned short tmp[16];
            *(uint4*)(tmp)     = *(const uint4*)(vp);
            *(uint4*)(tmp + 8) = *(const uint4*)(vp + 8);
#pragma unroll
            for (int i = 0; i < 16; ++i) Vt[(db + i) * 32 + kk] = tmp[i];
            if (ks + 1 < nsteps) {   // prefetch next V tile
                __builtin_prefetch(vp + (size_t)32 * stride, 0, 0);
            }
        }
        __syncthreads();

        if (kBase <= qRow0 + 15) {
            // ---- load ALL 8 K fragments first (distinct regs -> clause the
            //      16 global_load_b128s, then drain with WMMAs) ----
            v16bf kf[8];
#pragma unroll
            for (int c = 0; c < 4; ++c) {
                kf[c]     = load_frag_global(Kb + (size_t)(kBase)      * stride + c * 32, stride, lane);
                kf[c + 4] = load_frag_global(Kb + (size_t)(kBase + 16) * stride + c * 32, stride, lane);
            }
            v8f sc[2] = {};
#pragma unroll
            for (int c = 0; c < 4; ++c) {
                sc[0] = wmma_bf16(qfrag[c], kf[c],     sc[0]);
                sc[1] = wmma_bf16(qfrag[c], kf[c + 4], sc[1]);
            }
            // mask, scale, row max (reduce across each 16-lane half)
            float m_cur[8];
#pragma unroll
            for (int r = 0; r < 8; ++r) {
                const int row = qRow0 + r + 8 * halfl;
                float v0 = sc[0][r] * scale;
                float v1 = sc[1][r] * scale;
                if (kBase + lcol      > row) v0 = -1e30f;
                if (kBase + 16 + lcol > row) v1 = -1e30f;
                sc[0][r] = v0; sc[1][r] = v1;
                float m = fmaxf(v0, v1);
                m = fmaxf(m, __shfl_xor(m, 1, 32));
                m = fmaxf(m, __shfl_xor(m, 2, 32));
                m = fmaxf(m, __shfl_xor(m, 4, 32));
                m = fmaxf(m, __shfl_xor(m, 8, 32));
                m_cur[r] = m;
            }
            // online softmax update
            float resc[8];
#pragma unroll
            for (int r = 0; r < 8; ++r) {
                const float mnew = fmaxf(mrow[r], m_cur[r]);
                resc[r] = __expf(mrow[r] - mnew);
                mrow[r] = mnew;
                const float p0 = __expf(sc[0][r] - mnew);
                const float p1 = __expf(sc[1][r] - mnew);
                sc[0][r] = p0; sc[1][r] = p1;
                float s = p0 + p1;
                s += __shfl_xor(s, 1, 32);
                s += __shfl_xor(s, 2, 32);
                s += __shfl_xor(s, 4, 32);
                s += __shfl_xor(s, 8, 32);
                lrow[r] = lrow[r] * resc[r] + s;
            }
#pragma unroll
            for (int t = 0; t < 8; ++t)
#pragma unroll
                for (int r = 0; r < 8; ++r) acc[t][r] *= resc[r];

            // P (C layout) -> bf16 A fragment via wave-private LDS
#pragma unroll
            for (int r = 0; r < 8; ++r) {
                const int row = r + 8 * halfl;
                myP[row * 32 + lcol]      = f32_to_bf16(sc[0][r]);
                myP[row * 32 + 16 + lcol] = f32_to_bf16(sc[1][r]);
            }
            v16bf pfrag = load_frag_lds(myP, lane);

            // O += P @ V : 8 N-tiles of 16 over HD, fragments batched in pairs
#pragma unroll
            for (int t = 0; t < 8; t += 2) {
                v16bf vf0 = load_frag_lds(&Vt[((t + 0) * 16) * 32], lane);
                v16bf vf1 = load_frag_lds(&Vt[((t + 1) * 16) * 32], lane);
                acc[t + 0] = wmma_bf16(pfrag, vf0, acc[t + 0]);
                acc[t + 1] = wmma_bf16(pfrag, vf1, acc[t + 1]);
            }
        }
    }

    // normalize and store to out[b, s, h*HD + d] (bf16)
    unsigned short* ob = out + ((size_t)b * S) * (H * HD) + (size_t)h * HD;
#pragma unroll
    for (int t = 0; t < 8; ++t)
#pragma unroll
        for (int r = 0; r < 8; ++r) {
            const int row = qRow0 + r + 8 * halfl;
            const int col = t * 16 + lcol;
            ob[(size_t)row * (H * HD) + col] = f32_to_bf16(acc[t][r] / lrow[r]);
        }
}

// ---------------------------------------------------------------------------
// Launcher
// ---------------------------------------------------------------------------
extern "C" void kernel_launch(void* const* d_in, const int* in_sizes, int n_in,
                              void* d_out, int out_size, void* d_ws, size_t ws_size,
                              hipStream_t stream) {
    const float* x    = (const float*)d_in[0];
    const float* Wqkv = (const float*)d_in[1];
    const float* Wo   = (const float*)d_in[2];
    float* out = (float*)d_out;

    const int B = 2, S = 2048, D = 2048, H = 16, HD = 128;
    const int M  = B * S;        // 4096
    const int N1 = 3 * H * HD;   // 6144

    // workspace layout (bf16 elements)
    unsigned short* Xb   = (unsigned short*)d_ws;          // x as bf16      [M, D]
    unsigned short* WqT  = Xb  + (size_t)M  * D;           // Wqkv^T bf16    [N1, D]
    unsigned short* WoT  = WqT + (size_t)N1 * D;           // Wo^T bf16      [D, D]
    unsigned short* qkv  = WoT + (size_t)D  * D;           // qkv bf16       [M, N1]
    unsigned short* attn = qkv + (size_t)M  * N1;          // attn out bf16  [M, D]

    dim3 blk256(256);

    // 0) one-time precision/layout prep
    {
        const long n = (long)M * D;
        convert_f32_bf16<<<dim3((unsigned)(n / 4 / 256)), blk256, 0, stream>>>(x, Xb, n);
    }
    transpose_f32_bf16<<<dim3(N1 / 32, D / 32), dim3(32, 8), 0, stream>>>(Wqkv, WqT, D, N1);
    transpose_f32_bf16<<<dim3(D  / 32, D / 32), dim3(32, 8), 0, stream>>>(Wo,   WoT, D, D);

    // 1) QKV GEMM (bf16 WMMA, async-LDS double buffered)
    gemm_bf16<false><<<dim3(N1 / 128, M / 128), blk256, 0, stream>>>(
        Xb, WqT, qkv, M, N1, D);

    // 2) RoPE on q,k slices
    const long ropeThreads = (long)B * S * H * (HD / 2) * 2;
    rope_kernel<<<dim3((unsigned)((ropeThreads + 255) / 256)), blk256, 0, stream>>>(
        qkv, B, S, H, HD);

    // 3) causal flash attention
    attn_kernel<<<dim3(S / 128, H, B), blk256, 0, stream>>>(qkv, attn, S, H, HD);

    // 4) output projection (bf16 A/B, f32 out)
    gemm_bf16<true><<<dim3(D / 128, M / 128), blk256, 0, stream>>>(
        attn, WoT, out, M, D, H * HD);
}
